// DNC_38482906972754
// MI455X (gfx1250) — matrix operand order repaired and checked by reading, not compiled, not run on hardware
//
#include <hip/hip_runtime.h>
#include <hip/hip_bf16.h>

typedef __attribute__((ext_vector_type(16))) _Float16 v16h;
typedef __attribute__((ext_vector_type(8)))  float    v8f;

#define T_STEPS 128
#define BATCH   32
#define IN_DIM  128
#define OUT_DIM 128
#define N_SLOTS 256
#define M_DIM   64
#define R_HEADS 4
#define H_DIM   256
#define K_LSTM  640   // IN + M*R + H
#define NL      1024  // 4*H
#define KC_L    20    // 640/32
#define KC_S    8     // 256/32
#define LOSS_OFF ((size_t)T_STEPS*BATCH*OUT_DIM)

// ---------------- fragment loaders ----------------
// A operand (16x32 f16) from LDS, per CDNA5 ISA layout:
// lane<16: M=lane, K={0..7,16..23}; lane>=16: M=lane-16, K={8..15,24..31}
__device__ __forceinline__ v16h loadA(const _Float16* lds, int rowStride,
                                      int colOff, int kc, int lane) {
  const int row = lane & 15;
  const int laneHi = lane >> 4;
  const int base = row * rowStride + colOff + kc * 32 + laneHi * 8; // halves, even
  union { v16h h; unsigned u[8]; } fr;
  const unsigned* p = (const unsigned*)lds;
#pragma unroll
  for (int v = 0; v < 8; ++v) {
    const int k0 = (v < 4) ? (2 * v) : (16 + 2 * (v - 4));
    fr.u[v] = p[(base + k0) >> 1];
  }
  return fr.h;
}

// B operand (32x16 f16) pre-swizzled in global: lane-contiguous 32B chunks.
__device__ __forceinline__ v16h loadB(const _Float16* g, int nt, int KC,
                                      int kc, int lane) {
  union { v16h h; uint4 q[2]; } fr;
  const uint4* p = (const uint4*)(g + (((size_t)(nt * KC + kc) * 32 + lane) << 4));
  fr.q[0] = p[0];
  fr.q[1] = p[1];
  return fr.h;
}

__device__ __forceinline__ float sigm(float x) { return 1.f / (1.f + __expf(-x)); }
__device__ __forceinline__ float softplusf(float x) { return (x > 20.f) ? x : log1pf(__expf(x)); }

__device__ __forceinline__ float wredSum(float v) {
#pragma unroll
  for (int m = 16; m >= 1; m >>= 1) v += __shfl_xor(v, m, 32);
  return v;
}
__device__ __forceinline__ float wredMax(float v) {
#pragma unroll
  for (int m = 16; m >= 1; m >>= 1) v = fmaxf(v, __shfl_xor(v, m, 32));
  return v;
}

// ---------------- weight pack / f16 convert ----------------
__global__ void dnc_pack(const float* __restrict__ W_lstm, const float* __restrict__ W_pre,
                         const float* __restrict__ W_key, const float* __restrict__ W_teach,
                         const float* __restrict__ W_ro,
                         _Float16* __restrict__ packedL, _Float16* __restrict__ packedS,
                         float* __restrict__ wroSum, float* __restrict__ lossSlot) {
  const int total = 729088; // 655360 + 65536 + 8192
  for (int idx = blockIdx.x * blockDim.x + threadIdx.x; idx < total;
       idx += gridDim.x * blockDim.x) {
    if (idx == 0) *lossSlot = 0.f;
    if (idx < 655360) {                       // W_lstm -> WMMA-B swizzled f16
      int e = idx & 15, l = (idx >> 4) & 31;
      int kc = (idx >> 9) % 20, nt = idx / 10240;
      int v = e >> 1, hh = e & 1, laneHi = l >> 4;
      int col = nt * 16 + (l & 15);
      int k = kc * 32 + laneHi * 8 + ((v < 4) ? 2 * v : 16 + 2 * (v - 4)) + hh;
      packedL[idx] = (_Float16)W_lstm[k * NL + col];
    } else if (idx < 720896) {                // [W_pre|W_key|W_teach] -> swizzled f16
      int j = idx - 655360;
      int e = j & 15, l = (j >> 4) & 31;
      int kc = (j >> 9) & 7, nt = j / 4096;
      int v = e >> 1, hh = e & 1, laneHi = l >> 4;
      int col = nt * 16 + (l & 15);
      int k = kc * 32 + laneHi * 8 + ((v < 4) ? 2 * v : 16 + 2 * (v - 4)) + hh;
      float val;
      if (col < 128)      val = W_pre[k * 128 + col];
      else if (col < 192) val = W_key[k * 64 + (col - 128)];
      else                val = W_teach[k * 64 + (col - 192)];
      packedS[j] = (_Float16)val;
    } else {                                  // collapse W_ro over R broadcast heads
      int j = idx - 720896;
      int m = j >> 7, o = j & 127;
      float s = 0.f;
#pragma unroll
      for (int r = 0; r < 4; ++r) s += W_ro[(m * 4 + r) * 128 + o];
      wroSum[j] = s;
    }
  }
}

// ---------------- persistent per-batch-pair recurrence ----------------
// 16 blocks x 512 threads (16 waves). Block b handles batches {2b, 2b+1}.
__global__ void __launch_bounds__(512) dnc_main(
    const float* __restrict__ xs, const float* __restrict__ mem0,
    const float* __restrict__ b_lstm, const float* __restrict__ b_pre,
    const float* __restrict__ W_beta,
    const _Float16* __restrict__ packedL, const _Float16* __restrict__ packedS,
    const float* __restrict__ wroSum, float* __restrict__ out) {
  extern __shared__ char smem[];
  float*    memL   = (float*)(smem);                 // 2*256*64 f32  (131072 B)
  _Float16* actA   = (_Float16*)(smem + 131072);     // 16*640 f16    (20480 B)
  float*    cLDS   = (float*)(smem + 151552);        // 2*256
  float*    wLDS   = (float*)(smem + 153600);        // 2*256
  float*    preLDS = (float*)(smem + 155648);        // 2*128
  float*    kLDS   = (float*)(smem + 156672);        // 2*64
  float*    teachL = (float*)(smem + 157184);        // 2*64
  float*    readL  = (float*)(smem + 157696);        // 2*64
  float*    betaL  = (float*)(smem + 158208);        // 2
  float*    lossA  = (float*)(smem + 158272);        // 2

  const int tid = threadIdx.x, lane = tid & 31, wave = tid >> 5;
  const int bg0 = blockIdx.x * 2;

  // init: memory tile into LDS, zero activations/state
  for (int i = tid; i < 2 * N_SLOTS * M_DIM; i += 512)
    memL[i] = mem0[(size_t)bg0 * N_SLOTS * M_DIM + i];
  for (int i = tid; i < 16 * K_LSTM; i += 512) actA[i] = (_Float16)0.f;
  for (int i = tid; i < 2 * H_DIM; i += 512) cLDS[i] = 0.f;
  if (tid < 2) lossA[tid] = 0.f;
  __syncthreads();

  for (int t = 0; t < T_STEPS; ++t) {
    // (a) stage x_t into A-matrix rows 0..1 (f16)
    if (tid < 2 * IN_DIM) {
      int lb = tid >> 7, i = tid & 127;
      actA[lb * K_LSTM + i] =
          (_Float16)xs[((size_t)t * BATCH + bg0 + lb) * IN_DIM + i];
    }
    __syncthreads();

    // (b) LSTM GEMM: z = [x|read|h] @ W_lstm ; wave jt owns 16 cols of all 4 gates
    v8f ai = {}, af = {}, ag = {}, ao = {};
    {
      const int jt = wave;
#pragma unroll 1
      for (int kc = 0; kc < KC_L; ++kc) {
        if (kc + 1 < KC_L)
          __builtin_prefetch(packedL + (((size_t)(jt * KC_L + kc + 1) * 32 + lane) << 4), 0, 3);
        v16h a  = loadA(actA, K_LSTM, 0, kc, lane);
        v16h w0 = loadB(packedL, jt,      KC_L, kc, lane);
        v16h w1 = loadB(packedL, 16 + jt, KC_L, kc, lane);
        v16h w2 = loadB(packedL, 32 + jt, KC_L, kc, lane);
        v16h w3 = loadB(packedL, 48 + jt, KC_L, kc, lane);
        ai = __builtin_amdgcn_wmma_f32_16x16x32_f16(false, a, false, w0, (short)0, ai, false, false);
        af = __builtin_amdgcn_wmma_f32_16x16x32_f16(false, a, false, w1, (short)0, af, false, false);
        ag = __builtin_amdgcn_wmma_f32_16x16x32_f16(false, a, false, w2, (short)0, ag, false, false);
        ao = __builtin_amdgcn_wmma_f32_16x16x32_f16(false, a, false, w3, (short)0, ao, false, false);
      }
    }
    __syncthreads();

    // (c) LSTM nonlinearity directly in WMMA C layout; h -> actA (f16), c -> LDS
    {
      const int col = wave * 16 + (lane & 15);
      const int rowHi = (lane >> 4) * 8;
      const float bi = b_lstm[col], bf = b_lstm[H_DIM + col],
                  bg = b_lstm[2 * H_DIM + col], bo = b_lstm[3 * H_DIM + col];
#pragma unroll
      for (int v = 0; v < 8; ++v) {
        const int lb = rowHi + v;
        if (lb < 2) {
          float zi = ai[v] + bi, zf = af[v] + bf, zg = ag[v] + bg, zo = ao[v] + bo;
          float cold = cLDS[lb * H_DIM + col];
          float cn = sigm(zf) * cold + sigm(zi) * tanhf(zg);
          float hn = sigm(zo) * tanhf(cn);
          cLDS[lb * H_DIM + col] = cn;
          actA[lb * K_LSTM + IN_DIM + M_DIM * R_HEADS + col] = (_Float16)hn;
        }
      }
    }
    __syncthreads();

    // (d) head GEMM: h @ [W_pre|W_key|W_teach]  (one 16x16 tile per wave) + beta GEMV
    {
      const int nt = wave;
      v8f acc = {};
#pragma unroll 1
      for (int kc = 0; kc < KC_S; ++kc) {
        v16h a = loadA(actA, K_LSTM, IN_DIM + M_DIM * R_HEADS, kc, lane);
        v16h w = loadB(packedS, nt, KC_S, kc, lane);
        acc = __builtin_amdgcn_wmma_f32_16x16x32_f16(false, a, false, w, (short)0, acc, false, false);
      }
      const int col = nt * 16 + (lane & 15);
      const int rowHi = (lane >> 4) * 8;
#pragma unroll
      for (int v = 0; v < 8; ++v) {
        const int lb = rowHi + v;
        if (lb < 2) {
          float r = acc[v];
          if (col < OUT_DIM)              preLDS[lb * OUT_DIM + col] = r + b_pre[col];
          else if (col < OUT_DIM + M_DIM) kLDS[lb * M_DIM + (col - OUT_DIM)] = r;
          else                            teachL[lb * M_DIM + (col - OUT_DIM - M_DIM)] = r;
        }
      }
      if (wave == 0 && lane < 2) {
        float s = 0.f;
        for (int j = 0; j < H_DIM; ++j)
          s += (float)actA[lane * K_LSTM + IN_DIM + M_DIM * R_HEADS + j] * W_beta[j];
        betaL[lane] = softplusf(s);
      }
    }
    __syncthreads();

    // (e) memory attention: wave lb<2 owns batch lb; mem entirely in LDS
    if (wave < 2) {
      const int lb = wave;
      const float k0 = kLDS[lb * M_DIM + 2 * lane];
      const float k1 = kLDS[lb * M_DIM + 2 * lane + 1];
      float d0 = k0 - teachL[lb * M_DIM + 2 * lane];
      float d1 = k1 - teachL[lb * M_DIM + 2 * lane + 1];
      float ls = wredSum(d0 * d0 + d1 * d1);
      if (lane == 0) lossA[lb] += ls;
      const float nk = sqrtf(wredSum(k0 * k0 + k1 * k1));
      const float beta = betaL[lb];
      float sim[8];
#pragma unroll 1
      for (int g = 0; g < 8; ++g) {             // cosine sim, 8 slots/lane
        const int n = g * 32 + lane;
        const float* mp = memL + ((lb * N_SLOTS + n) << 6);
        float dot = 0.f, nr = 0.f;
        for (int m = 0; m < M_DIM; ++m) {
          float mv = mp[m];
          dot += mv * kLDS[lb * M_DIM + m];
          nr += mv * mv;
        }
        sim[g] = dot / fmaxf(sqrtf(nr) * nk, 1e-8f);
      }
      float mx = sim[0];
#pragma unroll
      for (int g = 1; g < 8; ++g) mx = fmaxf(mx, sim[g]);
      mx = wredMax(mx);
      float p[8], S = 0.f;
#pragma unroll
      for (int g = 0; g < 8; ++g) { p[g] = __expf(beta * (sim[g] - mx)); S += p[g]; }
      S = wredSum(S);
      const float inv = 1.f / S;
#pragma unroll
      for (int g = 0; g < 8; ++g) wLDS[lb * N_SLOTS + g * 32 + lane] = p[g] * inv;
      // additive write + read (post-update), lane owns 2 memory columns
      float ra0 = 0.f, ra1 = 0.f;
      float* mb = memL + ((lb * N_SLOTS) << 6) + 2 * lane;
#pragma unroll 1
      for (int n = 0; n < N_SLOTS; ++n) {
        const float w = wLDS[lb * N_SLOTS + n];
        float v0 = mb[n << 6] + w * k0;
        float v1 = mb[(n << 6) + 1] + w * k1;
        mb[n << 6] = v0; mb[(n << 6) + 1] = v1;
        ra0 += w * v0; ra1 += w * v1;
      }
      readL[lb * M_DIM + 2 * lane] = ra0;
      readL[lb * M_DIM + 2 * lane + 1] = ra1;
      // broadcast read into A-matrix (R identical heads, f16)
      _Float16* ar = actA + lb * K_LSTM + IN_DIM + (2 * lane) * R_HEADS;
      _Float16 h0 = (_Float16)ra0, h1 = (_Float16)ra1;
      ar[0] = h0; ar[1] = h0; ar[2] = h0; ar[3] = h0;
      ar[4] = h1; ar[5] = h1; ar[6] = h1; ar[7] = h1;
    }
    __syncthreads();

    // (f) out = pre_out + read @ Wro_sum  (R-collapsed)
    if (tid < 2 * OUT_DIM) {
      const int lb = tid >> 7, o = tid & 127;
      float acc = preLDS[lb * OUT_DIM + o];
      for (int m = 0; m < M_DIM; ++m)
        acc += readL[lb * M_DIM + m] * wroSum[m * OUT_DIM + o];
      out[((size_t)t * BATCH + bg0 + lb) * OUT_DIM + o] = acc;
    }
    // next (a) only touches actA x-region: no trailing barrier needed
  }
  __syncthreads();
  if (tid == 0)
    atomicAdd(out + LOSS_OFF, (lossA[0] + lossA[1]) / (float)(BATCH * M_DIM));
}

extern "C" void kernel_launch(void* const* d_in, const int* in_sizes, int n_in,
                              void* d_out, int out_size, void* d_ws, size_t ws_size,
                              hipStream_t stream) {
  (void)in_sizes; (void)n_in; (void)out_size; (void)ws_size;
  const float* xs      = (const float*)d_in[0];
  const float* mem0    = (const float*)d_in[1];
  const float* W_lstm  = (const float*)d_in[2];
  const float* b_lstm  = (const float*)d_in[3];
  const float* W_pre   = (const float*)d_in[4];
  const float* b_pre   = (const float*)d_in[5];
  const float* W_key   = (const float*)d_in[6];
  const float* W_beta  = (const float*)d_in[7];
  const float* W_teach = (const float*)d_in[8];
  const float* W_ro    = (const float*)d_in[9];
  float* out = (float*)d_out;
  char* ws = (char*)d_ws;
  _Float16* packedL = (_Float16*)ws;                    // 1,310,720 B
  _Float16* packedS = (_Float16*)(ws + 1310720);        //   131,072 B
  float*    wroSum  = (float*)(ws + 1310720 + 131072);  //    32,768 B

  static bool attrDone = false;
  if (!attrDone) {
    hipFuncSetAttribute((const void*)dnc_main,
                        hipFuncAttributeMaxDynamicSharedMemorySize, 158720);
    attrDone = true;
  }

  dnc_pack<<<1424, 512, 0, stream>>>(W_lstm, W_pre, W_key, W_teach, W_ro,
                                     packedL, packedS, wroSum, out + LOSS_OFF);
  dnc_main<<<16, 512, 158720, stream>>>(xs, mem0, b_lstm, b_pre, W_beta,
                                        packedL, packedS, wroSum, out);
}
